// ERMVP_66013647339868
// MI455X (gfx1250) — matrix operand order repaired and checked by reading, not compiled, not run on hardware
//
#include <hip/hip_runtime.h>
#include <hip/hip_bf16.h>

// ---------------- problem constants ----------------
#define N_   8
#define C_   256
#define H_   100
#define W_   252
#define HW_  (H_*W_)          // 25200
#define S_   2520             // tokens kept per CAV
#define K_   315              // clusters
#define SP_  2528             // S padded to 16 (158 tiles)
#define KNN_ 10
#define ROWT_ (HW_/16)        // 1575 row tiles for score GEMM
#define TPB_  (SP_/16)        // 158 tiles per dim for dist GEMM

typedef __bf16 v16bf __attribute__((ext_vector_type(16)));
typedef float  v8f   __attribute__((ext_vector_type(8)));

// ---------------- workspace layout (bytes) ----------------
static constexpr size_t al256(size_t x){ return (x + 255) & ~size_t(255); }
static constexpr size_t OFF_W1BF  = 0;
static constexpr size_t OFF_SCORE = al256(OFF_W1BF  + (size_t)C_*C_*2);
static constexpr size_t OFF_HIST  = al256(OFF_SCORE + (size_t)N_*HW_*4);
static constexpr size_t OFF_META  = al256(OFF_HIST  + (size_t)N_*4096*4);   // thresh[8],cntAbove[8],cntH[8],cntT[8]
static constexpr size_t OFF_TIDX  = al256(OFF_META  + 4*N_*4);
static constexpr size_t OFF_CONF  = al256(OFF_TIDX  + (size_t)N_*S_*4);
static constexpr size_t OFF_TOKF  = al256(OFF_CONF  + (size_t)N_*S_*4);
static constexpr size_t OFF_TOKB  = al256(OFF_TOKF  + (size_t)N_*SP_*C_*4);
static constexpr size_t OFF_SQN   = al256(OFF_TOKB  + (size_t)N_*SP_*C_*2);
static constexpr size_t OFF_DMAX  = al256(OFF_SQN   + (size_t)N_*SP_*4);
static constexpr size_t OFF_DENS  = al256(OFF_DMAX  + (size_t)N_*4);
static constexpr size_t OFF_CSC   = al256(OFF_DENS  + (size_t)N_*SP_*4);
static constexpr size_t OFF_IDOWN = al256(OFF_CSC   + (size_t)N_*SP_*4);
static constexpr size_t OFF_ICLU  = al256(OFF_IDOWN + (size_t)N_*K_*4);
static constexpr size_t OFF_ALLW  = al256(OFF_ICLU  + (size_t)N_*SP_*4);
static constexpr size_t OFF_MERG  = al256(OFF_ALLW  + (size_t)N_*K_*4);
static constexpr size_t OFF_DIST  = al256(OFF_MERG  + (size_t)N_*K_*C_*4);
// dist = N*SP*SP*4 = 204.5 MB at the tail

// ---------------- utility kernels ----------------
__global__ void ermvp_zero_kernel(float* p, size_t n) {
    size_t i = (size_t)blockIdx.x*256 + threadIdx.x;
    if (i < n) p[i] = 0.0f;
}

__global__ void ermvp_cvt_w1_kernel(const float* __restrict__ w1, __bf16* __restrict__ w1bf) {
    int i = blockIdx.x*256 + threadIdx.x;
    if (i < C_*C_) w1bf[i] = (__bf16)w1[i];
}

// ---------------- stage 1: fused score MLP (WMMA bf16) ----------------
// one wave computes 16 pixels x full 256 hidden channels, fused relu+dot(w2)+sigmoid
__global__ __launch_bounds__(128)
void ermvp_score_mlp_kernel(const float* __restrict__ x, const __bf16* __restrict__ w1bf,
                            const float* __restrict__ b1, const float* __restrict__ w2,
                            const float* __restrict__ b2, const float* __restrict__ pri,
                            float* __restrict__ score)
{
    const int lane = threadIdx.x & 31;
    const int wave = threadIdx.x >> 5;
    const int tile = blockIdx.x*4 + wave;          // 0 .. N*1575-1 (exact)
    const int n    = tile / ROWT_;
    const int row0 = (tile % ROWT_) * 16;
    const int lh   = lane >> 4;                    // lane half
    const int l15  = lane & 15;
    const int p    = row0 + l15;                   // A row = pixel
    const float* xc = x + (size_t)n*C_*HW_ + p;    // x[n, c, p] at xc[c*HW_]

    // preload all 8 A k-fragments (K=256 total), fp32 -> bf16 on the fly
    v16bf afr[8];
#pragma unroll
    for (int kt = 0; kt < 8; ++kt) {
        const int kb = kt*32 + lh*16;
#pragma unroll
        for (int j = 0; j < 8; ++j) {
            const int k = kb + 2*j;
            afr[kt][2*j]   = (__bf16)xc[(size_t)k*HW_];
            afr[kt][2*j+1] = (__bf16)xc[(size_t)(k+1)*HW_];
        }
    }

    float sp0=0.f,sp1=0.f,sp2=0.f,sp3=0.f,sp4=0.f,sp5=0.f,sp6=0.f,sp7=0.f;

#pragma unroll 1
    for (int nt = 0; nt < 16; ++nt) {
        const int d = nt*16 + l15;                     // B column = hidden channel
        const __bf16* wrow = w1bf + (size_t)d*C_ + lh*16;
        __builtin_prefetch(wrow + 16*C_);              // global_prefetch_b8 next tile
        v8f acc = {};
#pragma unroll
        for (int kt = 0; kt < 8; ++kt) {
            v16bf b;
#pragma unroll
            for (int j = 0; j < 8; ++j) {
                b[2*j]   = wrow[kt*32 + 2*j];
                b[2*j+1] = wrow[kt*32 + 2*j + 1];
            }
            acc = __builtin_amdgcn_wmma_f32_16x16x32_bf16(
                      false, afr[kt], false, b, (short)0, acc, false, false);
        }
        const float w2d = w2[d];
        const float b1d = b1[d];
        float h;
        h = acc[0]+b1d; sp0 += (h>0.f?h:0.f)*w2d;
        h = acc[1]+b1d; sp1 += (h>0.f?h:0.f)*w2d;
        h = acc[2]+b1d; sp2 += (h>0.f?h:0.f)*w2d;
        h = acc[3]+b1d; sp3 += (h>0.f?h:0.f)*w2d;
        h = acc[4]+b1d; sp4 += (h>0.f?h:0.f)*w2d;
        h = acc[5]+b1d; sp5 += (h>0.f?h:0.f)*w2d;
        h = acc[6]+b1d; sp6 += (h>0.f?h:0.f)*w2d;
        h = acc[7]+b1d; sp7 += (h>0.f?h:0.f)*w2d;
    }
    // reduce each row-sum across the 16 lanes of each half (D: lane = column)
    float sp[8] = {sp0,sp1,sp2,sp3,sp4,sp5,sp6,sp7};
#pragma unroll
    for (int r = 0; r < 8; ++r) {
        float v = sp[r];
        v += __shfl_xor(v, 1, 16);
        v += __shfl_xor(v, 2, 16);
        v += __shfl_xor(v, 4, 16);
        v += __shfl_xor(v, 8, 16);
        sp[r] = v;
    }
    if (l15 < 8) {
        float v = sp[0];
#pragma unroll
        for (int r = 1; r < 8; ++r) if (l15 == r) v = sp[r];
        const int pm = row0 + lh*8 + l15;             // D row = lh*8 + r
        float sc = 1.0f / (1.0f + __expf(-(v + b2[0])));
        sc *= pri[(size_t)n*HW_ + pm];
        score[(size_t)n*HW_ + pm] = sc;
    }
}

// ---------------- stage 2: radix top-k (S of HW) ----------------
__global__ void ermvp_hist_kernel(const float* __restrict__ score, unsigned* __restrict__ hist) {
    int q = blockIdx.x*256 + threadIdx.x;
    if (q >= N_*HW_) return;
    int n = q / HW_;
    unsigned bin = __float_as_uint(score[q]) >> 20;   // positive floats: bits monotonic
    atomicAdd(&hist[n*4096 + bin], 1u);
}

__global__ void ermvp_thresh_kernel(const unsigned* __restrict__ hist,
                                    int* __restrict__ thresh, int* __restrict__ cnt_above) {
    int n = blockIdx.x;
    if (threadIdx.x != 0) return;
    int cum = 0, tb = 0, ca = 0;
    for (int b = 4095; b >= 0; --b) {
        int h = (int)hist[n*4096 + b];
        if (cum + h >= S_) { tb = b; ca = cum; break; }
        cum += h;
    }
    thresh[n] = tb; cnt_above[n] = ca;
}

__global__ void ermvp_compact_kernel(const float* __restrict__ score,
                                     const int* __restrict__ thresh, const int* __restrict__ cnt_above,
                                     int* __restrict__ cntH, int* __restrict__ cntT,
                                     int* __restrict__ topk_idx, float* __restrict__ conf) {
    int q = blockIdx.x*256 + threadIdx.x;
    if (q >= N_*HW_) return;
    int n = q / HW_, p = q % HW_;
    float sc = score[q];
    int bin = (int)(__float_as_uint(sc) >> 20);
    int tb = thresh[n];
    if (bin > tb) {
        int pos = atomicAdd(&cntH[n], 1);
        topk_idx[n*S_+pos] = p; conf[n*S_+pos] = sc;
    } else if (bin == tb) {
        int pos = cnt_above[n] + atomicAdd(&cntT[n], 1);
        if (pos < S_) { topk_idx[n*S_+pos] = p; conf[n*S_+pos] = sc; }
    }
}

// ---------------- stage 3: gather + layernorm + conf scale ----------------
__global__ __launch_bounds__(256)
void ermvp_gather_ln_kernel(const float* __restrict__ x, const int* __restrict__ topk_idx,
                            const float* __restrict__ conf, float* __restrict__ tokens_f,
                            __bf16* __restrict__ tokens_bf, float* __restrict__ sqn)
{
    const int ns = blockIdx.x;              // N*SP
    const int n = ns / SP_, s = ns % SP_;
    const int c = threadIdx.x;
    __shared__ float red[256];
    float v = 0.f, cf = 0.f;
    if (s < S_) {
        int p = topk_idx[n*S_ + s];
        cf = conf[n*S_ + s];
        v = x[(size_t)n*C_*HW_ + (size_t)c*HW_ + p];
    }
    red[c] = v; __syncthreads();
    for (int st = 128; st > 0; st >>= 1) { if (c < st) red[c] += red[c+st]; __syncthreads(); }
    const float mu = red[0] * (1.0f/C_);
    __syncthreads();
    const float d = v - mu;
    red[c] = d*d; __syncthreads();
    for (int st = 128; st > 0; st >>= 1) { if (c < st) red[c] += red[c+st]; __syncthreads(); }
    const float var = red[0] * (1.0f/C_);
    __syncthreads();
    const float t = (s < S_) ? d * rsqrtf(var + 1e-5f) * cf : 0.0f;
    const size_t o = (size_t)n*SP_*C_ + (size_t)s*C_ + c;
    tokens_f[o]  = t;
    tokens_bf[o] = (__bf16)t;
    red[c] = t*t; __syncthreads();
    for (int st = 128; st > 0; st >>= 1) { if (c < st) red[c] += red[c+st]; __syncthreads(); }
    if (c == 0) sqn[n*SP_ + s] = red[0];
}

// ---------------- stage 4: pairwise distance via WMMA Gram matrix ----------------
__global__ __launch_bounds__(128)
void ermvp_dist_gemm_kernel(const __bf16* __restrict__ tokens_bf, const float* __restrict__ sqn,
                            float* __restrict__ dist, unsigned* __restrict__ distmax)
{
    const int lane = threadIdx.x & 31;
    const int wave = threadIdx.x >> 5;
    const int tile = blockIdx.x*4 + wave;            // N * 158 * 158 (exact)
    const int n  = tile / (TPB_*TPB_);
    const int t2 = tile % (TPB_*TPB_);
    const int i0 = (t2 / TPB_) * 16;
    const int j0 = (t2 % TPB_) * 16;
    const int lh = lane >> 4, l15 = lane & 15;
    const __bf16* tb   = tokens_bf + (size_t)n*SP_*C_;
    const __bf16* arow = tb + (size_t)(i0 + l15)*C_;   // A row i
    const __bf16* bcol = tb + (size_t)(j0 + l15)*C_;   // B column j (token j)

    v8f acc = {};
#pragma unroll 1
    for (int kt = 0; kt < 8; ++kt) {
        __builtin_prefetch(bcol + (kt+1)*32);
        v16bf a, b;
        const int kbA = kt*32 + lh*16;
        const int kbB = kt*32 + lh*16;
#pragma unroll
        for (int j = 0; j < 8; ++j) {
            const int ka = kbA + 2*j;
            a[2*j] = arow[ka];     a[2*j+1] = arow[ka+1];
            const int kb = kbB + 2*j;
            b[2*j] = bcol[kb];     b[2*j+1] = bcol[kb+1];
        }
        acc = __builtin_amdgcn_wmma_f32_16x16x32_bf16(
                  false, a, false, b, (short)0, acc, false, false);
    }
    const int j = j0 + l15;
    const float sj = sqn[n*SP_ + j];
    float lmax = 0.f;
#pragma unroll
    for (int r = 0; r < 8; ++r) {
        const int i = i0 + lh*8 + r;
        const float si = sqn[n*SP_ + i];
        float d2 = si + sj - 2.0f*acc[r];
        float dd = sqrtf(fmaxf(d2, 1e-12f)) * (1.0f/16.0f);   // / sqrt(C)
        if (j >= S_ || i >= S_) dd = 1e30f;
        else lmax = fmaxf(lmax, dd);
        dist[(size_t)n*SP_*SP_ + (size_t)i*SP_ + j] = dd;
    }
    for (int m = 16; m >= 1; m >>= 1) lmax = fmaxf(lmax, __shfl_xor(lmax, m, 32));
    if (lane == 0) atomicMax(distmax + n, __float_as_uint(lmax));
}

// ---------------- stage 5: KNN density (top-10 smallest per row) ----------------
__global__ __launch_bounds__(256)
void ermvp_density_kernel(const float* __restrict__ dist, float* __restrict__ density)
{
    const int ni = blockIdx.x;             // N*S
    const int n = ni / S_, i = ni % S_;
    const int t = threadIdx.x;
    const float* drow = dist + (size_t)n*SP_*SP_ + (size_t)i*SP_;
    float top[KNN_];
#pragma unroll
    for (int q = 0; q < KNN_; ++q) top[q] = 1e30f;
    for (int j = t; j < S_; j += 256) {
        const float d = drow[j];
        if (d < top[KNN_-1]) {
#pragma unroll
            for (int q = KNN_-1; q >= 0; --q) {
                const float a  = top[q];
                const float pr = (q > 0) ? top[q-1] : -1e30f;
                top[q] = (d >= a) ? a : ((d >= pr) ? d : pr);
            }
        }
    }
    __shared__ float cand[256*KNN_];
#pragma unroll
    for (int q = 0; q < KNN_; ++q) cand[t*KNN_+q] = top[q];
    __syncthreads();
    for (int st = 128; st > 0; st >>= 1) {
        if (t < st) {
            float out[KNN_];
            int ia = 0, ib = 0;
#pragma unroll
            for (int q = 0; q < KNN_; ++q) {
                const float va = cand[t*KNN_+ia], vb = cand[(t+st)*KNN_+ib];
                if (va <= vb) { out[q] = va; ++ia; } else { out[q] = vb; ++ib; }
            }
#pragma unroll
            for (int q = 0; q < KNN_; ++q) cand[t*KNN_+q] = out[q];
        }
        __syncthreads();
    }
    if (t == 0) {
        float s = 0.f;
#pragma unroll
        for (int q = 0; q < KNN_; ++q) { const float v = cand[q]; s += v*v; }
        float dens = __expf(-s * (1.0f/KNN_));
        unsigned h = (unsigned)ni * 2654435761u;     // deterministic tie-break noise
        dens += ((float)h * 2.3283064365386963e-10f) * 1e-6f;
        density[n*SP_ + i] = dens;
    }
}

// ---------------- stage 6: parent distance -> cluster score ----------------
__global__ __launch_bounds__(256)
void ermvp_parent_kernel(const float* __restrict__ dist, const float* __restrict__ density,
                         const unsigned* __restrict__ distmax, float* __restrict__ cscore)
{
    const int ni = blockIdx.x;             // N*S
    const int n = ni / S_, i = ni % S_;
    const int t = threadIdx.x;
    const float di = density[n*SP_ + i];
    const float* drow = dist + (size_t)n*SP_*SP_ + (size_t)i*SP_;
    const float* dens = density + n*SP_;
    float m = 1e30f;
    for (int j = t; j < S_; j += 256)
        if (dens[j] > di) m = fminf(m, drow[j]);
    __shared__ float red[256];
    red[t] = m; __syncthreads();
    for (int st = 128; st > 0; st >>= 1) { if (t < st) red[t] = fminf(red[t], red[t+st]); __syncthreads(); }
    if (t == 0) {
        const float dm = __uint_as_float(distmax[n]);
        cscore[n*SP_ + i] = fminf(red[0], dm) * di;
    }
}

// ---------------- stage 7: top-K cluster centers ----------------
__global__ __launch_bounds__(256)
void ermvp_topk_cluster_kernel(float* __restrict__ cscore, int* __restrict__ index_down)
{
    const int n = blockIdx.x, t = threadIdx.x;
    __shared__ float v[256]; __shared__ int ix[256];
    float* cs = cscore + n*SP_;
    for (int k = 0; k < K_; ++k) {
        float bm = -1e30f; int bi = 0;
        for (int j = t; j < S_; j += 256) { const float c = cs[j]; if (c > bm) { bm = c; bi = j; } }
        v[t] = bm; ix[t] = bi; __syncthreads();
        for (int st = 128; st > 0; st >>= 1) {
            if (t < st && v[t+st] > v[t]) { v[t] = v[t+st]; ix[t] = ix[t+st]; }
            __syncthreads();
        }
        if (t == 0) { index_down[n*K_ + k] = ix[0]; cs[ix[0]] = -1e30f; }
        __syncthreads();
    }
}

// ---------------- stage 8: cluster assignment ----------------
__global__ __launch_bounds__(256)
void ermvp_assign_kernel(const float* __restrict__ dist, const int* __restrict__ index_down,
                         int* __restrict__ idx_cluster)
{
    __shared__ int idl[K_];
    const int n = blockIdx.y;
    const int t = threadIdx.x;
    for (int q = t; q < K_; q += 256) idl[q] = index_down[n*K_ + q];
    __syncthreads();
    const int i = blockIdx.x*256 + t;
    if (i >= S_) return;
    const float* dbase = dist + (size_t)n*SP_*SP_;
    float bm = 1e30f; int bk = 0;
    for (int k = 0; k < K_; ++k) {
        const float d = dbase[(size_t)idl[k]*SP_ + i];
        if (d < bm) { bm = d; bk = k; }
    }
    idx_cluster[n*SP_ + i] = bk;
}

__global__ void ermvp_override_kernel(const int* __restrict__ index_down, int* __restrict__ idx_cluster) {
    const int q = blockIdx.x*256 + threadIdx.x;   // N*K
    if (q >= N_*K_) return;
    const int n = q / K_, k = q % K_;
    idx_cluster[n*SP_ + index_down[q]] = k;
}

// ---------------- stage 9: confidence-weighted merge ----------------
__global__ void ermvp_allw_kernel(const int* __restrict__ idx_cluster, const float* __restrict__ conf,
                                  float* __restrict__ all_w) {
    const int q = blockIdx.x*256 + threadIdx.x;   // N*S
    if (q >= N_*S_) return;
    const int n = q / S_, i = q % S_;
    atomicAdd(&all_w[n*K_ + idx_cluster[n*SP_ + i]], conf[q]);
}

__global__ __launch_bounds__(256)
void ermvp_merge_kernel(const float* __restrict__ tokens_f, const int* __restrict__ idx_cluster,
                        const float* __restrict__ conf, const float* __restrict__ all_w,
                        float* __restrict__ merged) {
    const int ni = blockIdx.x;                    // N*S
    const int n = ni / S_, i = ni % S_;
    const int c = threadIdx.x;
    const int k = idx_cluster[n*SP_ + i];
    const float w = conf[n*S_ + i] / (all_w[n*K_ + k] + 1e-6f);
    const float tv = tokens_f[(size_t)n*SP_*C_ + (size_t)i*C_ + c];
    atomicAdd(&merged[((size_t)n*K_ + k)*C_ + c], tv*w);
}

// ---------------- stage 10: scatter back + ego override ----------------
__global__ __launch_bounds__(256)
void ermvp_scatter_kernel(const float* __restrict__ merged, const int* __restrict__ idx_cluster,
                          const int* __restrict__ topk_idx, float* __restrict__ out) {
    const int ni = blockIdx.x;                    // N*S
    const int n = ni / S_, s = ni % S_;
    const int c = threadIdx.x;
    const int p = topk_idx[n*S_ + s];
    const int k = idx_cluster[n*SP_ + s];
    out[(size_t)n*C_*HW_ + (size_t)c*HW_ + p] = merged[((size_t)n*K_ + k)*C_ + c];
}

__global__ void ermvp_ego_kernel(const float* __restrict__ x, const int* __restrict__ record_len,
                                 int rcount, float* __restrict__ out) {
    const size_t i = (size_t)blockIdx.x*256 + threadIdx.x;
    if (i >= (size_t)N_*C_*HW_) return;
    const int n = (int)(i / ((size_t)C_*HW_));
    bool ego = false; int acc = 0;
    for (int r = 0; r < rcount; ++r) { if (n == acc) ego = true; acc += record_len[r]; }
    if (ego) out[i] = x[i];
}

// ---------------- launcher ----------------
extern "C" void kernel_launch(void* const* d_in, const int* in_sizes, int n_in,
                              void* d_out, int out_size, void* d_ws, size_t ws_size,
                              hipStream_t stream) {
    const float* x    = (const float*)d_in[0];
    const float* w1   = (const float*)d_in[1];
    const float* b1   = (const float*)d_in[2];
    const float* w2   = (const float*)d_in[3];
    const float* b2   = (const float*)d_in[4];
    const float* pri  = (const float*)d_in[5];
    const int*   rlen = (const int*)d_in[6];
    const int    rcount = in_sizes[6];
    float* out = (float*)d_out;

    char* ws = (char*)d_ws;
    __bf16*   w1bf    = (__bf16*)(ws + OFF_W1BF);
    float*    score   = (float*)(ws + OFF_SCORE);
    unsigned* hist    = (unsigned*)(ws + OFF_HIST);
    int*      thresh  = (int*)(ws + OFF_META);
    int*      cntAb   = thresh + N_;
    int*      cntH    = thresh + 2*N_;
    int*      cntT    = thresh + 3*N_;
    int*      tidx    = (int*)(ws + OFF_TIDX);
    float*    conf    = (float*)(ws + OFF_CONF);
    float*    tokf    = (float*)(ws + OFF_TOKF);
    __bf16*   tokb    = (__bf16*)(ws + OFF_TOKB);
    float*    sqn     = (float*)(ws + OFF_SQN);
    unsigned* dmax    = (unsigned*)(ws + OFF_DMAX);
    float*    dens    = (float*)(ws + OFF_DENS);
    float*    csc     = (float*)(ws + OFF_CSC);
    int*      idown   = (int*)(ws + OFF_IDOWN);
    int*      iclu    = (int*)(ws + OFF_ICLU);
    float*    allw    = (float*)(ws + OFF_ALLW);
    float*    merged  = (float*)(ws + OFF_MERG);
    float*    dist    = (float*)(ws + OFF_DIST);

    auto zblk = [](size_t n){ return (unsigned)((n + 255) / 256); };

    // init scratch that is accumulated into
    ermvp_zero_kernel<<<zblk((size_t)N_*4096), 256, 0, stream>>>((float*)hist, (size_t)N_*4096);
    ermvp_zero_kernel<<<1, 256, 0, stream>>>((float*)thresh, 4*N_);
    ermvp_zero_kernel<<<zblk((size_t)N_*K_), 256, 0, stream>>>(allw, (size_t)N_*K_);
    ermvp_zero_kernel<<<zblk((size_t)N_*K_*C_), 256, 0, stream>>>(merged, (size_t)N_*K_*C_);
    ermvp_zero_kernel<<<1, 256, 0, stream>>>((float*)dmax, N_);
    ermvp_zero_kernel<<<zblk((size_t)N_*C_*HW_), 256, 0, stream>>>(out, (size_t)N_*C_*HW_);

    // stage 1: score MLP (WMMA)
    ermvp_cvt_w1_kernel<<<zblk((size_t)C_*C_), 256, 0, stream>>>(w1, w1bf);
    ermvp_score_mlp_kernel<<<N_*ROWT_/4, 128, 0, stream>>>(x, w1bf, b1, w2, b2, pri, score);

    // stage 2: radix top-k select
    ermvp_hist_kernel<<<zblk((size_t)N_*HW_), 256, 0, stream>>>(score, hist);
    ermvp_thresh_kernel<<<N_, 64, 0, stream>>>(hist, thresh, cntAb);
    ermvp_compact_kernel<<<zblk((size_t)N_*HW_), 256, 0, stream>>>(score, thresh, cntAb, cntH, cntT, tidx, conf);

    // stage 3: gather + layernorm
    ermvp_gather_ln_kernel<<<N_*SP_, 256, 0, stream>>>(x, tidx, conf, tokf, tokb, sqn);

    // stage 4: pairwise distance (WMMA Gram)
    ermvp_dist_gemm_kernel<<<N_*TPB_*TPB_/4, 128, 0, stream>>>(tokb, sqn, dist, dmax);

    // stages 5-8: DPC-KNN clustering
    ermvp_density_kernel<<<N_*S_, 256, 0, stream>>>(dist, dens);
    ermvp_parent_kernel<<<N_*S_, 256, 0, stream>>>(dist, dens, dmax, csc);
    ermvp_topk_cluster_kernel<<<N_, 256, 0, stream>>>(csc, idown);
    {
        dim3 g((S_ + 255)/256, N_);
        ermvp_assign_kernel<<<g, 256, 0, stream>>>(dist, idown, iclu);
    }
    ermvp_override_kernel<<<zblk((size_t)N_*K_), 256, 0, stream>>>(idown, iclu);

    // stage 9: merge
    ermvp_allw_kernel<<<zblk((size_t)N_*S_), 256, 0, stream>>>(iclu, conf, allw);
    ermvp_merge_kernel<<<N_*S_, 256, 0, stream>>>(tokf, iclu, conf, allw, merged);

    // stage 10: scatter + ego
    ermvp_scatter_kernel<<<N_*S_, 256, 0, stream>>>(merged, iclu, tidx, out);
    ermvp_ego_kernel<<<zblk((size_t)N_*C_*HW_), 256, 0, stream>>>(x, rlen, rcount, out);
}
